// dfg_block_67697274520363
// MI455X (gfx1250) — matrix operand order repaired and checked by reading, not compiled, not run on hardware
//
#include <hip/hip_runtime.h>
#include <hip/hip_bf16.h>

#define NN 50000
#define HH 256
#define BB 50
#define NPG 1000
#define KSEL 500

typedef __attribute__((ext_vector_type(16))) __bf16 v16bf;
typedef __attribute__((ext_vector_type(8)))  __bf16 bf8v;
typedef __attribute__((ext_vector_type(8)))  float  v8f;

// Native L2-side f32 atomic add (GLOBAL_ATOMIC_ADD_F32, no CAS loop).
__device__ __forceinline__ void atomAddF(float* p, float v) {
    unsafeAtomicAdd(p, v);
}

// ---------------- degree / normalization ----------------
__global__ void k_deg_init(float* __restrict__ deg) {
    int i = blockIdx.x * blockDim.x + threadIdx.x;
    if (i < NN) deg[i] = 1.0f;               // self-loop
}
__global__ void k_deg_count(const int* __restrict__ dst, float* __restrict__ deg, int E) {
    int e = blockIdx.x * blockDim.x + threadIdx.x;
    if (e < E) atomAddF(&deg[dst[e]], 1.0f);
}
__global__ void k_deg_finish(float* __restrict__ deg) {
    int i = blockIdx.x * blockDim.x + threadIdx.x;
    if (i < NN) deg[i] = rsqrtf(deg[i]);     // deg >= 1 always
}

// ---------------- layer 1: one-hot gather + self-loop init ----------------
__global__ void k_gather1(const int* __restrict__ x, const float* __restrict__ W1,
                          const float* __restrict__ dinv, float* __restrict__ xw1,
                          float* __restrict__ out1) {
    int idx = blockIdx.x * blockDim.x + threadIdx.x;   // exactly N*256 threads
    int i = idx >> 8, h = idx & 255;
    float w  = W1[x[i] * HH + h];
    float di = dinv[i];
    xw1[idx]  = w;
    out1[idx] = w * di * di;                 // self-loop term
}

// ---------------- normalized edge scatter (32 thr / edge, 8 feats each) ----------
__global__ void k_scatter(const float* __restrict__ xw, const float* __restrict__ dinv,
                          const int* __restrict__ src, const int* __restrict__ dst,
                          float* __restrict__ out, int E) {
    int t = blockIdx.x * blockDim.x + threadIdx.x;
    int e = t >> 5;
    if (e >= E) return;
    int c = t & 31;
    int s = src[e], d = dst[e];
    float nrm = dinv[s] * dinv[d];
    const float4* xs = (const float4*)(xw + ((size_t)s << 8) + (c << 3));
    float* od = out + ((size_t)d << 8) + (c << 3);
    float4 v0 = xs[0], v1 = xs[1];
    atomAddF(od + 0, v0.x * nrm); atomAddF(od + 1, v0.y * nrm);
    atomAddF(od + 2, v0.z * nrm); atomAddF(od + 3, v0.w * nrm);
    atomAddF(od + 4, v1.x * nrm); atomAddF(od + 5, v1.y * nrm);
    atomAddF(od + 6, v1.z * nrm); atomAddF(od + 7, v1.w * nrm);
}

// ---------------- bias + relu -> bf16 (layer-1 output feeds WMMA) ----------------
__global__ void k_relu_bf16(const float* __restrict__ acc, const float* __restrict__ b,
                            __bf16* __restrict__ h1) {
    int idx = blockIdx.x * blockDim.x + threadIdx.x;
    int h = idx & 255;
    float v = acc[idx] + b[h];
    h1[idx] = (__bf16)(v > 0.f ? v : 0.f);
}

// ---------------- W2 -> bf16 transposed ([n][k] row-major) ----------------
__global__ void k_w2t(const float* __restrict__ W2, __bf16* __restrict__ w2t) {
    int idx = blockIdx.x * blockDim.x + threadIdx.x;   // 65536 threads
    int n = idx >> 8, k = idx & 255;
    w2t[idx] = (__bf16)W2[k * HH + n];
}

// ---------------- bf16 WMMA GEMM with fused layer-2 self-loop init ----------------
// D[50000x256] = A * B; also writes Dinit = D * dinv[row]^2 (saves a full pass).
// 8 waves/block; block covers 16 rows x 128 cols; grid = (50000/16, 2)
__global__ void k_gemm(const __bf16* __restrict__ A, const __bf16* __restrict__ Bt,
                       const float* __restrict__ dinv,
                       float* __restrict__ D, float* __restrict__ Dinit) {
    int wave = threadIdx.x >> 5;
    int lane = threadIdx.x & 31;
    int halfsel = lane >> 4;                 // 0: lanes 0-15, 1: lanes 16-31
    int mt = blockIdx.x;                     // 16-row tile
    int nt = blockIdx.y * 8 + wave;          // 16-col tile (0..15)
    int row  = mt * 16 + (lane & 15);        // A row this lane owns
    int colB = nt * 16 + (lane & 15);        // B column this lane owns
    const __bf16* arow = A  + (size_t)row  * HH;
    const __bf16* brow = Bt + (size_t)colB * HH;   // W2^T row = B column
    v8f c = {};
    #pragma unroll
    for (int kk = 0; kk < HH; kk += 32) {
        // A 16x32 frag: K chunks [halfsel*8, +8) and [16+halfsel*8, +8)
        bf8v alo = *(const bf8v*)(arow + kk + halfsel * 8);
        bf8v ahi = *(const bf8v*)(arow + kk + 16 + halfsel * 8);
        v16bf a = __builtin_shufflevector(alo, ahi,
                    0,1,2,3,4,5,6,7,8,9,10,11,12,13,14,15);
        // B 32x16 frag: K chunks [halfsel*8, +8) and [16+halfsel*8, +8)
        bf8v blo = *(const bf8v*)(brow + kk + halfsel * 8);
        bf8v bhi = *(const bf8v*)(brow + kk + 16 + halfsel * 8);
        v16bf b = __builtin_shufflevector(blo, bhi,
                    0,1,2,3,4,5,6,7,8,9,10,11,12,13,14,15);
        c = __builtin_amdgcn_wmma_f32_16x16x32_bf16(
                false, a, false, b, (short)0, c, false, false);
    }
    // C/D layout: VGPR r -> M = r + 8*halfsel, N = lane&15
    int drow = mt * 16 + halfsel * 8;
    int dcol = nt * 16 + (lane & 15);
    #pragma unroll
    for (int r = 0; r < 8; ++r) {
        size_t o = (size_t)(drow + r) * HH + dcol;
        float di = dinv[drow + r];
        D[o]     = c[r];
        Dinit[o] = c[r] * di * di;
    }
}

// ---------------- fused layer-2 bias+relu (in place) + scorer dots -------------
// 1 wave per node: h2 = relu(acc + b2); trel = h2.Wrel; scr = h2.Wroot + brel
__global__ void k_relu2_score(float* __restrict__ acc, const float* __restrict__ b2,
                              const float* __restrict__ Wrel, const float* __restrict__ Wroot,
                              const float* __restrict__ brel,
                              float* __restrict__ trel, float* __restrict__ scr) {
    int node = blockIdx.x * 8 + (threadIdx.x >> 5);  // grid 6250 * 8 = 50000 exact
    int lane = threadIdx.x & 31;
    float* h = acc + (size_t)node * HH;
    float s1 = 0.f, s2 = 0.f;
    #pragma unroll
    for (int j = 0; j < 8; ++j) {
        int k = lane + 32 * j;
        float v = h[k] + b2[k];
        v = v > 0.f ? v : 0.f;
        h[k] = v;
        s1 += v * Wrel[k];
        s2 += v * Wroot[k];
    }
    #pragma unroll
    for (int o = 16; o > 0; o >>= 1) {
        s1 += __shfl_xor(s1, o, 32);
        s2 += __shfl_xor(s2, o, 32);
    }
    if (lane == 0) { trel[node] = s1; scr[node] = s2 + brel[0]; }
}
__global__ void k_score_scatter(const float* __restrict__ trel, const int* __restrict__ src,
                                const int* __restrict__ dst, float* __restrict__ scr, int E) {
    int e = blockIdx.x * blockDim.x + threadIdx.x;
    if (e < E) atomAddF(&scr[dst[e]], trel[src[e]]);
}

// ---------------- per-graph top-K (bitonic in LDS) + tanh-scaled max-pool -------
__global__ void k_topk_pool(const float* __restrict__ scr, const float* __restrict__ h2,
                            float* __restrict__ out) {
    __shared__ float sval[1024];
    __shared__ int   sidx[1024];
    int b = blockIdx.x, t = threadIdx.x;
    const float NEG = -3.0e38f;
    if (t < NPG) { sval[t] = scr[b * NPG + t]; sidx[t] = t; }
    else         { sval[t] = NEG;              sidx[t] = 0; }
    __syncthreads();
    // bitonic sort, descending
    for (int k = 2; k <= 1024; k <<= 1) {
        for (int j = k >> 1; j > 0; j >>= 1) {
            int ixj = t ^ j;
            if (ixj > t) {
                bool up = ((t & k) == 0);
                float v0 = sval[t], v1 = sval[ixj];
                bool sw = up ? (v0 < v1) : (v0 > v1);
                if (sw) {
                    sval[t] = v1; sval[ixj] = v0;
                    int ti = sidx[t]; sidx[t] = sidx[ixj]; sidx[ixj] = ti;
                }
            }
            __syncthreads();
        }
    }
    if (t < KSEL) sval[t] = tanhf(sval[t]);   // SAGPool nonlinearity
    __syncthreads();
    if (t < HH) {
        float m = NEG;
        for (int k2 = 0; k2 < KSEL; ++k2) {
            int node = b * NPG + sidx[k2];
            m = fmaxf(m, h2[(size_t)node * HH + t] * sval[k2]);
        }
        out[b * HH + t] = m;
    }
}

extern "C" void kernel_launch(void* const* d_in, const int* in_sizes, int n_in,
                              void* d_out, int out_size, void* d_ws, size_t ws_size,
                              hipStream_t stream) {
    const int*   x     = (const int*)d_in[0];
    const int*   ei    = (const int*)d_in[1];
    const float* W1    = (const float*)d_in[3];
    const float* b1    = (const float*)d_in[4];
    const float* W2    = (const float*)d_in[5];
    const float* b2    = (const float*)d_in[6];
    const float* Wrel  = (const float*)d_in[7];
    const float* brel  = (const float*)d_in[8];
    const float* Wroot = (const float*)d_in[9];
    int E = in_sizes[1] / 2;
    const int* src = ei;
    const int* dst = ei + E;

    char* p = (char*)d_ws;
    auto carve = [&](size_t bytes) {
        void* q = (void*)p;
        p += (bytes + 255) & ~(size_t)255;
        return q;
    };
    float*  bufA = (float*) carve((size_t)NN * HH * 4);  // xw1, later out2/h2
    float*  bufB = (float*) carve((size_t)NN * HH * 4);  // out1, later xw2
    __bf16* h1   = (__bf16*)carve((size_t)NN * HH * 2);
    __bf16* w2t  = (__bf16*)carve((size_t)HH * HH * 2);
    float*  dinv = (float*) carve((size_t)NN * 4);
    float*  trel = (float*) carve((size_t)NN * 4);
    float*  scr  = (float*) carve((size_t)NN * 4);

    // degrees -> rsqrt
    k_deg_init  <<<(NN + 255) / 256, 256, 0, stream>>>(dinv);
    k_deg_count <<<(E + 255) / 256, 256, 0, stream>>>(dst, dinv, E);
    k_deg_finish<<<(NN + 255) / 256, 256, 0, stream>>>(dinv);

    // layer 1: gather rows of W1, scatter-aggregate, relu -> bf16
    k_gather1<<<NN, 256, 0, stream>>>(x, W1, dinv, bufA, bufB);
    k_scatter<<<((size_t)E * 32 + 255) / 256, 256, 0, stream>>>(bufA, dinv, src, dst, bufB, E);
    k_relu_bf16<<<NN, 256, 0, stream>>>(bufB, b1, h1);

    // layer 2: bf16 WMMA GEMM (fused self-loop init), scatter, fused relu+score
    k_w2t<<<HH, 256, 0, stream>>>(W2, w2t);
    dim3 gg(NN / 16, 2);
    k_gemm<<<gg, 256, 0, stream>>>(h1, w2t, dinv, bufB, bufA);
    k_scatter<<<((size_t)E * 32 + 255) / 256, 256, 0, stream>>>(bufB, dinv, src, dst, bufA, E);
    k_relu2_score<<<NN / 8, 256, 0, stream>>>(bufA, b2, Wrel, Wroot, brel, trel, scr);
    k_score_scatter<<<(E + 255) / 256, 256, 0, stream>>>(trel, src, dst, scr, E);

    // per-graph top-K + tanh-weighted global max pool
    k_topk_pool<<<BB, 1024, 0, stream>>>(scr, bufA, (float*)d_out);
}
